// GATModel_84275848282632
// MI455X (gfx1250) — compile-verified, matrix operand order checked by reading, not run on hardware
//
#include <hip/hip_runtime.h>
#include <hip/hip_bf16.h>
#include <stddef.h>
#include <stdint.h>

typedef __attribute__((ext_vector_type(16))) _Float16 v16h;
typedef __attribute__((ext_vector_type(8)))  _Float16 v8h;
typedef __attribute__((ext_vector_type(8)))  float    v8f;
typedef __attribute__((ext_vector_type(4)))  int      v4i;

#define GB 128   // batch
#define GN 256   // nodes
#define GD 128   // input dim
#define GF 64    // per-head features
#define GH 4     // heads
#define GR 256   // readout dim
#define GC 2     // classes
#define GM (GB * GN)        // 32768 rows
#define HF (GH * GF)        // 256

// ---------------------------------------------------------------------------
// CDNA5 async global->LDS copy (ASYNCcnt-tracked).
// Builtin signature (from hipcc diagnostic): (AS1 v4i*, AS3 v4i*, int, int)
__device__ __forceinline__ void async_b128(const void* g, void* l) {
#if __has_builtin(__builtin_amdgcn_global_load_async_to_lds_b128)
    __builtin_amdgcn_global_load_async_to_lds_b128(
        (__attribute__((address_space(1))) v4i*)g,
        (__attribute__((address_space(3))) v4i*)l, 0, 0);
#else
    unsigned lo = (unsigned)(unsigned long long)(uintptr_t)l;  // LDS byte address
    asm volatile("global_load_async_to_lds_b128 %0, %1, off"
                 :: "v"(lo), "v"((unsigned long long)(uintptr_t)g)
                 : "memory");
#endif
}

__device__ __forceinline__ void async_wait0() {
#if __has_builtin(__builtin_amdgcn_s_wait_asynccnt)
    __builtin_amdgcn_s_wait_asynccnt(0);
#else
    asm volatile("s_wait_asynccnt 0x0" ::: "memory");
#endif
}

// ---------------------------------------------------------------------------
// elementwise f32 -> f16
__global__ void k_cvt_f16(const float* __restrict__ src, _Float16* __restrict__ dst, int n) {
    int i = blockIdx.x * blockDim.x + threadIdx.x;
    if (i < n) dst[i] = (_Float16)src[i];
}

// transpose + convert: src f32 [K][Nn] row-major -> dst f16 [Nn][K]
__global__ void k_tcvt_f16(const float* __restrict__ src, _Float16* __restrict__ dst,
                           int K, int Nn) {
    int i = blockIdx.x * blockDim.x + threadIdx.x;
    if (i >= K * Nn) return;
    int r = i / K, k = i % K;
    dst[i] = (_Float16)src[k * Nn + r];
}

// Build Wt f16 [H][F][Din] (transposed per head) and wa f32 [H][2][Din] = W @ a halves.
__global__ void k_prep_w(const float* __restrict__ w, const float* __restrict__ a,
                         _Float16* __restrict__ wt, float* __restrict__ wa, int Din) {
    int i = blockIdx.x * blockDim.x + threadIdx.x;
    if (i >= GH * Din) return;
    int h = i / Din, k = i % Din;
    const float* wr = w + ((size_t)h * Din + k) * GF;
    const float* ar = a + h * (2 * GF);
    float a0 = 0.f, a1 = 0.f;
    for (int f = 0; f < GF; ++f) {
        float wv = wr[f];
        wt[((size_t)h * GF + f) * Din + k] = (_Float16)wv;
        a0 += wv * ar[f];
        a1 += wv * ar[GF + f];
    }
    wa[((size_t)h * 2 + 0) * Din + k] = a0;
    wa[((size_t)h * 2 + 1) * Din + k] = a1;
}

// s_src[h][m] = X[m,:] . wa[h][0], s_dst[h][m] = X[m,:] . wa[h][1]
__global__ void k_scores(const float* __restrict__ X, const float* __restrict__ wa,
                         float* __restrict__ s_src, float* __restrict__ s_dst, int Din) {
    __shared__ float wal[GH * 2 * 256];
    for (int t = threadIdx.x; t < GH * 2 * Din; t += blockDim.x) wal[t] = wa[t];
    __syncthreads();
    int m = blockIdx.x * blockDim.x + threadIdx.x;
    if (m >= GM) return;
    const float* xr = X + (size_t)m * Din;
    float acc[GH][2] = {};
    for (int k = 0; k < Din; ++k) {
        float xv = xr[k];
#pragma unroll
        for (int h = 0; h < GH; ++h) {
            acc[h][0] += xv * wal[(h * 2 + 0) * Din + k];
            acc[h][1] += xv * wal[(h * 2 + 1) * Din + k];
        }
    }
#pragma unroll
    for (int h = 0; h < GH; ++h) {
        s_src[(size_t)h * GM + m] = acc[h][0];
        s_dst[(size_t)h * GM + m] = acc[h][1];
    }
}

// wh_t[h][b][f][n] = (X @ W_h)[n,f] via WMMA, transposed f16 output.
// 64-row M-tile per block, one f-strip + 4 M-subtile accumulators per wave.
// grid (M/64, H), block 128.
__global__ void k_gemm_wh(const _Float16* __restrict__ Xh, const _Float16* __restrict__ Wt,
                          _Float16* __restrict__ wht, int Din) {
    int wave = threadIdx.x >> 5, lane = threadIdx.x & 31;
    int half = lane >> 4, lm = lane & 15;
    int h = blockIdx.y;
    int m0 = blockIdx.x * 64;
    int f0 = wave * 16;
    const _Float16* brow = Wt + ((size_t)h * GF + f0 + lm) * Din + 16 * half;
    const _Float16* arow = Xh + (size_t)(m0 + lm) * Din + 16 * half;
    size_t astep = (size_t)16 * Din;
    v8f acc0 = {}, acc1 = {}, acc2 = {}, acc3 = {};
    for (int k0 = 0; k0 < Din; k0 += 32) {
        v16h bb = *(const v16h*)(brow + k0);
        v16h a0 = *(const v16h*)(arow + k0);
        v16h a1 = *(const v16h*)(arow + astep + k0);
        v16h a2 = *(const v16h*)(arow + 2 * astep + k0);
        v16h a3 = *(const v16h*)(arow + 3 * astep + k0);
        acc0 = __builtin_amdgcn_wmma_f32_16x16x32_f16(false, a0, false, bb, (short)0, acc0, false, false);
        acc1 = __builtin_amdgcn_wmma_f32_16x16x32_f16(false, a1, false, bb, (short)0, acc1, false, false);
        acc2 = __builtin_amdgcn_wmma_f32_16x16x32_f16(false, a2, false, bb, (short)0, acc2, false, false);
        acc3 = __builtin_amdgcn_wmma_f32_16x16x32_f16(false, a3, false, bb, (short)0, acc3, false, false);
    }
    v8f* accs[4] = {&acc0, &acc1, &acc2, &acc3};
#pragma unroll
    for (int t = 0; t < 4; ++t) {
        int mt = m0 + t * 16;
        int b = mt / GN, nbase = mt % GN;
        _Float16* dst = wht + (((size_t)h * GB + b) * GF + f0 + lm) * GN + nbase + 8 * half;
        v8h o;
#pragma unroll
        for (int r = 0; r < 8; ++r) o[r] = (_Float16)(*accs[t])[r];
        *(v8h*)dst = o;
    }
}

// Fused attention: masked leaky-relu softmax + coeffs @ wh (WMMA).
// wh slab (64x256 f16 = 32KB) staged to LDS with async loads overlapped with softmax.
// grid (N/16, B, H), block 128.  mode 0 = ELU epilogue, 1 = identity.
__global__ void k_attn(const float* __restrict__ s_src, const float* __restrict__ s_dst,
                       const _Float16* __restrict__ wht, const float* __restrict__ adj,
                       float* __restrict__ out, int mode) {
    __shared__ __align__(32) _Float16 cf[16][GN];    // exp coeffs (unnormalized)
    __shared__ __align__(32) _Float16 whl[GF][GN];   // staged wh slab
    __shared__ float rinv[16];

    int wave = threadIdx.x >> 5, lane = threadIdx.x & 31;
    int i0 = blockIdx.x * 16, b = blockIdx.y, h = blockIdx.z;

    // kick off async staging of the whole wh[64][256] slab for this (h,b)
    {
        const _Float16* gsrc = wht + (((size_t)h * GB + b) * GF) * GN;
        _Float16* lbase = &whl[0][0];
        for (int c = threadIdx.x; c < (GF * GN) / 8; c += 128)   // 8 halfs per b128
            async_b128(gsrc + (size_t)c * 8, lbase + (size_t)c * 8);
    }

    const float* sd = s_dst + ((size_t)h * GB + b) * GN;

    // phase 1 (overlaps the async fetch): 4 rows per wave -> exp coeffs + 1/sum
    for (int q = 0; q < 4; ++q) {
        int row = wave * 4 + q;
        int i = i0 + row;
        float ss = s_src[((size_t)h * GB + b) * GN + i];
        const float* adjrow = adj + ((size_t)b * GN + i) * GN;
        float mx = -3.4e38f;
        for (int jc = 0; jc < GN; jc += 32) {
            int j = jc + lane;
            float sc = ss + sd[j];
            sc = sc > 0.f ? sc : 0.2f * sc;          // leaky relu (slope 0.2)
            sc = adjrow[j] > 0.f ? sc : -9e15f;      // mask
            mx = fmaxf(mx, sc);
        }
#pragma unroll
        for (int o = 16; o > 0; o >>= 1) mx = fmaxf(mx, __shfl_xor(mx, o, 32));
        float sum = 0.f;
        for (int jc = 0; jc < GN; jc += 32) {        // recompute (adj row is L0-hot)
            int j = jc + lane;
            float sc = ss + sd[j];
            sc = sc > 0.f ? sc : 0.2f * sc;
            sc = adjrow[j] > 0.f ? sc : -9e15f;
            float e = __expf(sc - mx);
            sum += e;
            cf[row][j] = (_Float16)e;
        }
#pragma unroll
        for (int o = 16; o > 0; o >>= 1) sum += __shfl_xor(sum, o, 32);
        if (lane == 0) rinv[row] = 1.0f / sum;
    }
    async_wait0();          // own async copies to LDS complete
    __syncthreads();        // everyone's copies + coeffs visible

    // phase 2: out[16x64] = coeffs[16x256] @ wh[256x64], all operands in LDS
    int half = lane >> 4, lm = lane & 15;
    int f0 = wave * 16;
    const _Float16* brow = &whl[f0 + lm][0] + 16 * half;
    v8f acc = {};
    for (int k0 = 0; k0 < GN; k0 += 32) {
        v16h aa = *(const v16h*)&cf[lm][k0 + 16 * half];
        v16h bb = *(const v16h*)(brow + k0);
        acc = __builtin_amdgcn_wmma_f32_16x16x32_f16(false, aa, false, bb,
                                                     (short)0, acc, false, false);
    }
    float* orow = out + ((size_t)b * GN + i0) * HF + (size_t)h * GF + f0 + lm;
#pragma unroll
    for (int r = 0; r < 8; ++r) {
        int ml = r + 8 * half;
        float v = acc[r] * rinv[ml];
        if (mode == 0) v = v > 0.f ? v : (__expf(v) - 1.f);   // ELU
        orow[(size_t)ml * HF] = v;
    }
}

// Generic WMMA GEMM: out f16 [M][Nout] = act(X[M][K] @ Wt[Nout][K]^T + bias)
// 64-row M-tile per block, one n-strip + 4 M-subtile accumulators per wave.
// grid (M/64, Nout/64), block 128. act: 0 none, 1 sigmoid.
__global__ void k_gemm(const _Float16* __restrict__ X, const _Float16* __restrict__ Wt,
                       const float* __restrict__ bias, _Float16* __restrict__ out,
                       int K, int Nout, int act) {
    int wave = threadIdx.x >> 5, lane = threadIdx.x & 31;
    int half = lane >> 4, lm = lane & 15;
    int m0 = blockIdx.x * 64;
    int n0 = blockIdx.y * 64 + wave * 16;
    const _Float16* arow = X + (size_t)(m0 + lm) * K + 16 * half;
    const _Float16* brow = Wt + (size_t)(n0 + lm) * K + 16 * half;
    size_t astep = (size_t)16 * K;
    v8f acc0 = {}, acc1 = {}, acc2 = {}, acc3 = {};
    for (int k0 = 0; k0 < K; k0 += 32) {
        v16h bb = *(const v16h*)(brow + k0);
        v16h a0 = *(const v16h*)(arow + k0);
        v16h a1 = *(const v16h*)(arow + astep + k0);
        v16h a2 = *(const v16h*)(arow + 2 * astep + k0);
        v16h a3 = *(const v16h*)(arow + 3 * astep + k0);
        acc0 = __builtin_amdgcn_wmma_f32_16x16x32_f16(false, a0, false, bb, (short)0, acc0, false, false);
        acc1 = __builtin_amdgcn_wmma_f32_16x16x32_f16(false, a1, false, bb, (short)0, acc1, false, false);
        acc2 = __builtin_amdgcn_wmma_f32_16x16x32_f16(false, a2, false, bb, (short)0, acc2, false, false);
        acc3 = __builtin_amdgcn_wmma_f32_16x16x32_f16(false, a3, false, bb, (short)0, acc3, false, false);
    }
    float bn = bias[n0 + lm];
    v8f* accs[4] = {&acc0, &acc1, &acc2, &acc3};
#pragma unroll
    for (int t = 0; t < 4; ++t) {
        _Float16* op = out + (size_t)(m0 + t * 16) * Nout + n0 + lm;
#pragma unroll
        for (int r = 0; r < 8; ++r) {
            int ml = r + 8 * half;
            float v = (*accs[t])[r] + bn;
            if (act == 1) v = 1.f / (1.f + __expf(-v));
            op[(size_t)ml * Nout] = (_Float16)v;
        }
    }
}

// h3 = mean over heads of layer-3 output; also place h3 into Xg[:, :64]
__global__ void k_build_h3(const float* __restrict__ hbuf, _Float16* __restrict__ h3h,
                           _Float16* __restrict__ xgh) {
    int i = blockIdx.x * blockDim.x + threadIdx.x;
    if (i >= GM * GF) return;
    int m = i >> 6, f = i & 63;
    size_t base = (size_t)m * HF + f;
    float v = 0.25f * (hbuf[base] + hbuf[base + GF] + hbuf[base + 2 * GF] + hbuf[base + 3 * GF]);
    h3h[(size_t)m * GF + f] = (_Float16)v;
    xgh[(size_t)m * (GF + GD) + f] = (_Float16)v;
}

// Xg[:, 64:192] = nodes (f16)
__global__ void k_build_xg(const float* __restrict__ nodes, _Float16* __restrict__ xgh) {
    int i = blockIdx.x * blockDim.x + threadIdx.x;
    if (i >= GM * GD) return;
    int m = i >> 7, d = i & 127;
    xgh[(size_t)m * (GF + GD) + GF + d] = (_Float16)nodes[(size_t)m * GD + d];
}

// readout[b][r] = sum_n gate[b,n,r] * proj[b,n,r]
__global__ void k_readout(const _Float16* __restrict__ gate, const _Float16* __restrict__ proj,
                          float* __restrict__ ro) {
    int b = blockIdx.x, r = threadIdx.x;
    float acc = 0.f;
    for (int n = 0; n < GN; ++n) {
        size_t ix = ((size_t)b * GN + n) * GR + r;
        acc += (float)gate[ix] * (float)proj[ix];
    }
    ro[(size_t)b * GR + r] = acc;
}

// logits[b][c] = readout[b] . wc[:,c] + bc[c]
__global__ void k_logits(const float* __restrict__ ro, const float* __restrict__ wc,
                         const float* __restrict__ bc, float* __restrict__ out) {
    int t = blockIdx.x * blockDim.x + threadIdx.x;
    if (t >= GB * GC) return;
    int b = t >> 1, c = t & 1;
    float acc = bc[c];
    for (int r = 0; r < GR; ++r) acc += ro[(size_t)b * GR + r] * wc[r * GC + c];
    out[t] = acc;
}

// ---------------------------------------------------------------------------
extern "C" void kernel_launch(void* const* d_in, const int* in_sizes, int n_in,
                              void* d_out, int out_size, void* d_ws, size_t ws_size,
                              hipStream_t stream) {
    const float* nodes = (const float*)d_in[0];
    const float* adj   = (const float*)d_in[1];
    const float* w1 = (const float*)d_in[2];
    const float* a1 = (const float*)d_in[3];
    const float* w2 = (const float*)d_in[4];
    const float* a2 = (const float*)d_in[5];
    const float* w3 = (const float*)d_in[6];
    const float* a3 = (const float*)d_in[7];
    const float* wi = (const float*)d_in[8];
    const float* bi = (const float*)d_in[9];
    const float* wj = (const float*)d_in[10];
    const float* bj = (const float*)d_in[11];
    const float* wc = (const float*)d_in[12];
    const float* bc = (const float*)d_in[13];
    float* logits = (float*)d_out;

    char* ws = (char*)d_ws;
    size_t off = 0;
    auto take = [&](size_t bytes) { char* p = ws + off; off += (bytes + 255) & ~(size_t)255; return p; };
    _Float16* Xh    = (_Float16*)take((size_t)GM * 256 * 2);
    _Float16* Wt    = (_Float16*)take((size_t)GH * GF * 256 * 2);
    float*    wa    = (float*)   take((size_t)GH * 2 * 256 * 4);
    float*    s_src = (float*)   take((size_t)GH * GM * 4);
    float*    s_dst = (float*)   take((size_t)GH * GM * 4);
    _Float16* wht   = (_Float16*)take((size_t)GH * GB * GF * GN * 2);
    float*    hbuf  = (float*)   take((size_t)GM * HF * 4);
    _Float16* h3h   = (_Float16*)take((size_t)GM * GF * 2);
    _Float16* xgh   = (_Float16*)take((size_t)GM * (GF + GD) * 2);
    _Float16* wit   = (_Float16*)take((size_t)GR * (GF + GD) * 2);
    _Float16* wjt   = (_Float16*)take((size_t)GR * GF * 2);
    _Float16* gateb = (_Float16*)take((size_t)GM * GR * 2);
    _Float16* projb = (_Float16*)take((size_t)GM * GR * 2);
    float*    ro    = (float*)   take((size_t)GB * GR * 4);

    const float* lw[3] = {w1, w2, w3};
    const float* la[3] = {a1, a2, a3};
    const int   din[3] = {GD, HF, HF};
    const float* lx[3] = {nodes, hbuf, hbuf};

    for (int l = 0; l < 3; ++l) {
        int Din = din[l];
        int ncvt = GM * Din;
        k_cvt_f16<<<(ncvt + 255) / 256, 256, 0, stream>>>(lx[l], Xh, ncvt);
        k_prep_w<<<(GH * Din + 255) / 256, 256, 0, stream>>>(lw[l], la[l], Wt, wa, Din);
        k_scores<<<GM / 256, 256, 0, stream>>>(lx[l], wa, s_src, s_dst, Din);
        k_gemm_wh<<<dim3(GM / 64, GH), 128, 0, stream>>>(Xh, Wt, wht, Din);
        k_attn<<<dim3(GN / 16, GB, GH), 128, 0, stream>>>(s_src, s_dst, wht, adj, hbuf,
                                                          l == 2 ? 1 : 0);
    }

    k_build_h3<<<(GM * GF + 255) / 256, 256, 0, stream>>>(hbuf, h3h, xgh);
    k_build_xg<<<(GM * GD + 255) / 256, 256, 0, stream>>>(nodes, xgh);
    k_tcvt_f16<<<((GF + GD) * GR + 255) / 256, 256, 0, stream>>>(wi, wit, GF + GD, GR);
    k_tcvt_f16<<<(GF * GR + 255) / 256, 256, 0, stream>>>(wj, wjt, GF, GR);
    k_gemm<<<dim3(GM / 64, GR / 64), 128, 0, stream>>>(xgh, wit, bi, gateb, GF + GD, GR, 1);
    k_gemm<<<dim3(GM / 64, GR / 64), 128, 0, stream>>>(h3h, wjt, bj, projb, GF, GR, 0);
    k_readout<<<GB, GR, 0, stream>>>(gateb, projb, ro);
    k_logits<<<1, GB * GC, 0, stream>>>(ro, wc, bc, logits);
    (void)in_sizes; (void)n_in; (void)out_size; (void)ws_size;
}
